// SpectralConvS2_11742440587397
// MI455X (gfx1250) — compile-verified, matrix operand order checked by reading, not run on hardware
//
#include <hip/hip_runtime.h>
#include <hip/hip_bf16.h>
#include <cstdint>

// ---------------------------------------------------------------------------
// SpectralConvS2 for MI455X (gfx1250, wave32, WMMA f32_16x16x32_f16)
// Sizes: B=32, C=O=128, NLAT=H=96, NLON=W=192, M=97, L=96, K=4656, N=B*C=4096
// ---------------------------------------------------------------------------

#define HID   128
#define NB    32
#define NLAT  96
#define NLON  192
#define MM    97           // MMAX
#define LL    96           // LMAX
#define NN    4096         // B*HID
#define KTOT  4656         // tril count

typedef _Float16 h16 __attribute__((ext_vector_type(16)));
typedef _Float16 h8  __attribute__((ext_vector_type(8)));
typedef float    f8  __attribute__((ext_vector_type(8)));
typedef int      v4i __attribute__((ext_vector_type(4)));

#define TWO_PI 6.2831853071795864769f

#define AS1 __attribute__((address_space(1)))
#define AS3 __attribute__((address_space(3)))

#if defined(__HIP_DEVICE_COMPILE__) && __has_builtin(__builtin_amdgcn_global_load_async_to_lds_b128)
#define HAVE_ASYNC_LDS 1
#else
#define HAVE_ASYNC_LDS 0
#endif

__device__ __forceinline__ void wait_async0() {
#if defined(__HIP_DEVICE_COMPILE__) && __has_builtin(__builtin_amdgcn_s_wait_asynccnt)
    __builtin_amdgcn_s_wait_asynccnt(0);
#else
    asm volatile("s_wait_asynccnt 0x0" ::: "memory");
#endif
}

// ---- WMMA helpers ----------------------------------------------------------
__device__ __forceinline__ f8 wmma16(h16 a, h16 b, f8 c) {
    // D = A(16x32) * B(32x16) + C, f32 accumulate
    return __builtin_amdgcn_wmma_f32_16x16x32_f16(false, a, false, b,
                                                  (short)0, c, false, false);
}

// Load an A/B fragment from LDS, row-major [16 rows][ldk K-elems] tile.
// CDNA5 16-bit operand layout (ISA 7.12.2): lane L holds row (L&15); its 16
// elements are K = kb..kb+7 and K = 16+kb..16+kb+7 with kb = (L>=16 ? 8 : 0).
__device__ __forceinline__ h16 frag_ld(const _Float16* base, int ldk,
                                       int row0, int k0) {
    const int lane = threadIdx.x & 31;
    const _Float16* p = base + (row0 + (lane & 15)) * ldk + k0 + ((lane >> 4) << 3);
    h8 lo = *(const h8*)(p);
    h8 hi = *(const h8*)(p + 16);
    return __builtin_shufflevector(lo, hi, 0,1,2,3,4,5,6,7,8,9,10,11,12,13,14,15);
}

// ---------------------------------------------------------------------------
// Kernel 1: rDFT over longitude as GEMM.
// XF[c2][m][n][h] = sum_w F[c2][m][w] * x[n][h][w],  F = {cos,-sin}(2pi m w/W)/W
// grid(NN, 7 m-tiles, 2), block 192 (6 waves; wave = h-tile)
// ---------------------------------------------------------------------------
__global__ void k1_dft(const float* __restrict__ x, float* __restrict__ xf) {
    extern __shared__ char smem[];
    _Float16* sX = (_Float16*)smem;                      // [96 h][192 w]  (B: [N][K])
    _Float16* sF = (_Float16*)(smem + NLAT*NLON*2);      // [16 m][192 w]  (A: [M][K])
    const int n  = blockIdx.x;
    const int mt = blockIdx.y;
    const int c2 = blockIdx.z;
    const int tid = threadIdx.x;

    const float* xr = x + (size_t)n * (NLAT * NLON);
    for (int idx = tid; idx < NLAT*NLON; idx += 192)
        sX[idx] = (_Float16)xr[idx];

    const float inv = 1.0f / (float)NLON;
    for (int idx = tid; idx < 16*NLON; idx += 192) {
        int r = idx / NLON, wc = idx % NLON;
        int m = mt*16 + r;
        float v = 0.0f;
        if (m < MM) {
            int ph = (m * wc) % NLON;                    // exact phase reduction
            float th = TWO_PI * (float)ph * inv;
            v = (c2 == 0 ? __cosf(th) : -__sinf(th)) * inv;
        }
        sF[idx] = (_Float16)v;
    }
    __syncthreads();

    const int wave = tid >> 5, lane = tid & 31;
    f8 acc = {};
    for (int kc = 0; kc < NLON; kc += 32) {
        h16 a = frag_ld(sF, NLON, 0, kc);
        h16 b = frag_ld(sX, NLON, wave*16, kc);
        acc = wmma16(a, b, acc);
    }
    const int hcol  = wave*16 + (lane & 15);
    const int rbase = (lane >> 4) << 3;
    if (mt < 6) {                                        // full tile, no bounds check
        for (int r = 0; r < 8; ++r) {
            int m = mt*16 + r + rbase;
            xf[(((size_t)c2*MM + m)*NN + n)*NLAT + hcol] = acc[r];
        }
    } else if (rbase == 0) {                             // tile 6: only m==96 valid
        xf[(((size_t)c2*MM + 96)*NN + n)*NLAT + hcol] = acc[0];
    }
}

// ---------------------------------------------------------------------------
// Kernel 2: forward Legendre per m:  XG[c2][k(l,m)][n] = sum_h Ahat[l][h]*XF[c2][m][n][h]
// Ahat = 2*pi * pct[l][m][h] * quad_w[h].  grid(64 n-blocks, 97, 2), block 256.
// ---------------------------------------------------------------------------
__global__ void k2_legfwd(const float* __restrict__ xf, const float* __restrict__ pct,
                          const float* __restrict__ qw, float* __restrict__ xg) {
    extern __shared__ char smem[];
    _Float16* sA = (_Float16*)smem;                      // [96 lrow][96 h]
    _Float16* sB = (_Float16*)(smem + 96*96*2);          // [64 nl][96 h]
    const int nb = blockIdx.x, m = blockIdx.y, c2 = blockIdx.z;
    const int tid = threadIdx.x;

    for (int idx = tid; idx < 96*96; idx += 256) {
        int lrow = idx / 96, h = idx % 96;
        int l = m + lrow;
        float v = 0.0f;
        if (l < LL) v = TWO_PI * pct[((size_t)l*MM + m)*NLAT + h] * qw[h];
        sA[idx] = (_Float16)v;
    }
    const float* xfm = xf + ((size_t)c2*MM + m)*NN*NLAT + (size_t)nb*64*NLAT;
    for (int idx = tid; idx < 64*96; idx += 256)
        sB[idx] = (_Float16)xfm[idx];
    __syncthreads();

    const int wave = tid >> 5, lane = tid & 31;
    const int nt = wave & 3, lt0 = wave >> 2;
    const int ltiles = (LL - m + 15) >> 4;               // tiles with any valid row
    const int lfull  = (LL - m) >> 4;                    // fully-valid tiles
    f8 acc[3] = {};
    for (int kc = 0; kc < 96; kc += 32) {
        h16 b = frag_ld(sB, 96, nt*16, kc);
        for (int j = 0; j < 3; ++j) {
            int lt = lt0 + 2*j;
            if (lt < ltiles) {
                h16 a = frag_ld(sA, 96, lt*16, kc);
                acc[j] = wmma16(a, b, acc[j]);
            }
        }
    }
    const int ncol  = nb*64 + nt*16 + (lane & 15);
    const int rbase = (lane >> 4) << 3;
    float* xgc = xg + (size_t)c2 * KTOT * NN;
    for (int j = 0; j < 3; ++j) {
        int lt = lt0 + 2*j;
        if (lt >= ltiles) continue;
        if (lt < lfull) {                                // full tile: unguarded
            for (int r = 0; r < 8; ++r) {
                int l = m + lt*16 + r + rbase;
                int k = (l*(l+1))/2 + m;                 // np.tril_indices order
                xgc[(size_t)k*NN + ncol] = acc[j][r];
            }
        } else {                                         // tail tile
            for (int r = 0; r < 8; ++r) {
                int l = m + lt*16 + r + rbase;
                if (l < LL) {
                    int k = (l*(l+1))/2 + m;
                    xgc[(size_t)k*NN + ncol] = acc[j][r];
                }
            }
        }
    }
}

// ---------------------------------------------------------------------------
// Kernel 3: per-mode complex channel mix. 2 modes per workgroup so each lane
// streams one 16B (2 modes x re,im) vector of w per (o,i) ([o][i][k][2], k fastest).
// w chunks are staged raw (f32) into LDS with GLOBAL_LOAD_ASYNC_TO_LDS_B128
// (ASYNCcnt) and the next chunk's async fill overlaps the WMMA compute; a
// synchronous fallback is used if the builtin is unavailable.
// out[b][o] = sum_i xg[b][i]*wc[o][i] (complex). grid(KTOT/2), block 512 (16 waves).
// ---------------------------------------------------------------------------
#define K3_SB_OFF  0                 // f16 [(kk,c2)][128 o][32 i] : 32768 B
#define K3_SA_OFF  32768             // f16 [(kk,c2)][32 b][32 i]  :  8192 B
#define K3_RAW_OFF 40960             // f32 raw w chunk            : 65536 B
#define K3_LDS     (K3_RAW_OFF + 65536)

__global__ void k3_modegemm(const float* __restrict__ xg, const float* __restrict__ w,
                            float* __restrict__ ym) {
    extern __shared__ char smem[];
    _Float16* sB   = (_Float16*)(smem + K3_SB_OFF);
    _Float16* sA   = (_Float16*)(smem + K3_SA_OFF);
    float*    rawW = (float*)(smem + K3_RAW_OFF);
    (void)rawW;
    const int k0  = blockIdx.x * 2;
    const int tid = threadIdx.x;
    const int wave = tid >> 5, lane = tid & 31;
    const int kk = wave >> 3;        // local mode 0..1
    const int ot = wave & 7;         // o-tile 0..7

#if HAVE_ASYNC_LDS
    // prologue: async-stage w chunk ic=0 (one 16B vector per lane per op)
    for (int p = tid; p < 128*32; p += 512) {
        int o = p >> 5, il = p & 31;
        const float* src = w + ((size_t)(o*HID + il)*KTOT + k0)*2;
        __builtin_amdgcn_global_load_async_to_lds_b128(
            (AS1 v4i*)src, (AS3 v4i*)(smem + K3_RAW_OFF + p*16), 0, 0);
    }
#endif

    f8 cre[2] = {}; f8 cim[2] = {};
    for (int ic = 0; ic < HID; ic += 32) {
#if HAVE_ASYNC_LDS
        wait_async0();                                   // chunk `ic` landed in LDS
        __syncthreads();
        // convert raw f32 -> f16 operand layout
        for (int p = tid; p < 128*32; p += 512) {
            int o = p >> 5, il = p & 31;
            float4 f0 = *(const float4*)(rawW + p*4);    // (k0,re)(k0,im)(k1,re)(k1,im)
            sB[(0*128 + o)*32 + il] = (_Float16)f0.x;
            sB[(1*128 + o)*32 + il] = (_Float16)f0.y;
            sB[(2*128 + o)*32 + il] = (_Float16)f0.z;
            sB[(3*128 + o)*32 + il] = (_Float16)f0.w;
        }
#else
        __syncthreads();
        for (int p = tid; p < 128*32; p += 512) {
            int o = p >> 5, il = p & 31;
            const float* src = w + ((size_t)(o*HID + ic + il)*KTOT + k0)*2;
            float4 f0 = *(const float4*)src;
            sB[(0*128 + o)*32 + il] = (_Float16)f0.x;
            sB[(1*128 + o)*32 + il] = (_Float16)f0.y;
            sB[(2*128 + o)*32 + il] = (_Float16)f0.z;
            sB[(3*128 + o)*32 + il] = (_Float16)f0.w;
        }
#endif
        // stage xg: rows (c2,k,b) x 32 i (direct, small)
        {
            int row = tid >> 2, q = tid & 3;             // 128 rows, 8 floats each
            int c2 = row >> 6, krow = (row >> 5) & 1, b = row & 31;
            const float* src = xg + ((size_t)c2*KTOT + k0 + krow)*NN + b*HID + ic + q*8;
            _Float16* dst = sA + ((krow*2 + c2)*32 + b)*32 + q*8;
            for (int j = 0; j < 8; ++j) dst[j] = (_Float16)src[j];
        }
        __syncthreads();
#if HAVE_ASYNC_LDS
        if (ic + 32 < HID) {                             // overlap next chunk with WMMA
            for (int p = tid; p < 128*32; p += 512) {
                int o = p >> 5, il = p & 31;
                const float* src = w + ((size_t)(o*HID + ic + 32 + il)*KTOT + k0)*2;
                __builtin_amdgcn_global_load_async_to_lds_b128(
                    (AS1 v4i*)src, (AS3 v4i*)(smem + K3_RAW_OFF + p*16), 0, 0);
            }
        }
#endif
        const _Float16* Are = sA + (kk*2 + 0)*1024;
        const _Float16* Aim = sA + (kk*2 + 1)*1024;
        const _Float16* Bre = sB + (kk*2 + 0)*4096;
        const _Float16* Bim = sB + (kk*2 + 1)*4096;
        h16 bre = frag_ld(Bre, 32, ot*16, 0);
        h16 bim = frag_ld(Bim, 32, ot*16, 0);
        for (int bt = 0; bt < 2; ++bt) {
            h16 are = frag_ld(Are, 32, bt*16, 0);
            h16 aim = frag_ld(Aim, 32, bt*16, 0);
            h16 ain = -aim;                              // f16 WMMA has no A-neg (ISA NEG=CNeg only)
            cre[bt] = wmma16(are, bre, cre[bt]);
            cre[bt] = wmma16(ain, bim, cre[bt]);
            cim[bt] = wmma16(are, bim, cim[bt]);
            cim[bt] = wmma16(aim, bre, cim[bt]);
        }
    }
    const int k = k0 + kk;
    const int o = ot*16 + (lane & 15);
    const int rbase = (lane >> 4) << 3;
    for (int bt = 0; bt < 2; ++bt)
        for (int r = 0; r < 8; ++r) {
            int brow = bt*16 + r + rbase;
            size_t n2 = (size_t)brow*HID + o;
            ym[(size_t)k*NN + n2]          = cre[bt][r];
            ym[((size_t)KTOT + k)*NN + n2] = cim[bt][r];
        }
}

// ---------------------------------------------------------------------------
// Kernel 4: inverse Legendre per m: XH[c2][m][n][h] = sum_l pct[l][m][h]*YM[c2][k(l,m)][n]
// grid(64 n-blocks, 97, 2), block 256. m=96 writes zeros (no tril modes).
// ---------------------------------------------------------------------------
__global__ void k4_leginv(const float* __restrict__ ym, const float* __restrict__ pct,
                          float* __restrict__ xh) {
    extern __shared__ char smem[];
    _Float16* sA = (_Float16*)smem;                      // [64 nl][96 lrow]
    _Float16* sB = (_Float16*)(smem + 64*96*2);          // [96 h][96 lrow]
    const int nb = blockIdx.x, m = blockIdx.y, c2 = blockIdx.z;
    const int tid = threadIdx.x;

    for (int idx = tid; idx < 96*64; idx += 256) {
        int lrow = idx >> 6, nl = idx & 63;
        int l = m + lrow;
        float v = 0.0f;
        if (l < LL) {
            int k = (l*(l+1))/2 + m;
            v = ym[((size_t)c2*KTOT + k)*NN + nb*64 + nl];
        }
        sA[nl*96 + lrow] = (_Float16)v;
    }
    for (int idx = tid; idx < 96*96; idx += 256) {
        int lrow = idx / 96, h = idx % 96;
        int l = m + lrow;
        float v = 0.0f;
        if (l < LL) v = pct[((size_t)l*MM + m)*NLAT + h];
        sB[h*96 + lrow] = (_Float16)v;
    }
    __syncthreads();

    const int wave = tid >> 5, lane = tid & 31;
    const int nt = wave & 3, ht0 = wave >> 2;
    f8 acc[3] = {};
    for (int kc = 0; kc < 96; kc += 32) {
        h16 a = frag_ld(sA, 96, nt*16, kc);
        for (int j = 0; j < 3; ++j) {
            h16 b = frag_ld(sB, 96, (ht0 + 2*j)*16, kc);
            acc[j] = wmma16(a, b, acc[j]);
        }
    }
    const int rbase = (lane >> 4) << 3;
    for (int j = 0; j < 3; ++j) {
        int h = (ht0 + 2*j)*16 + (lane & 15);
        for (int r = 0; r < 8; ++r) {
            int n = nb*64 + nt*16 + r + rbase;
            xh[(((size_t)c2*MM + m)*NN + n)*NLAT + h] = acc[j][r];
        }
    }
}

// ---------------------------------------------------------------------------
// Kernel 5: irfft * W as GEMM over mc = (m,c2) pairs (K padded 194->224):
// y[n][h][w] = sum_m cm*(XHre*cos - XHim*sin), cm = 1 for m=0,96 else 2.
// grid(NN, 2 h-halves, 2 w-halves), block 192 (6 waves).
// ---------------------------------------------------------------------------
__global__ void k5_idft(const float* __restrict__ xh, float* __restrict__ y) {
    extern __shared__ char smem[];
    _Float16* sA = (_Float16*)smem;                      // [48 h][224 mc]
    _Float16* sB = (_Float16*)(smem + 48*224*2);         // [96 w][224 mc]
    const int n  = blockIdx.x;
    const int hh = blockIdx.y;                           // h half
    const int wh = blockIdx.z;                           // w half
    const int tid = threadIdx.x;

    for (int idx = tid; idx < 224*48; idx += 192) {
        int m2 = idx / 48, hl = idx % 48;
        float v = 0.0f;
        if (m2 < 194) {
            int c2 = m2 & 1, m = m2 >> 1;
            v = xh[(((size_t)c2*MM + m)*NN + n)*NLAT + hh*48 + hl];
        }
        sA[hl*224 + m2] = (_Float16)v;
    }
    const float invW = 1.0f / (float)NLON;
    for (int idx = tid; idx < 96*224; idx += 192) {
        int wl = idx / 224, kidx = idx % 224;
        float v = 0.0f;
        if (kidx < 194) {
            int m = kidx >> 1, c2 = kidx & 1;
            int ww = wh*96 + wl;
            int ph = (m * ww) % NLON;
            float th = TWO_PI * (float)ph * invW;
            float cm = (m == 0 || m == MM-1) ? 1.0f : 2.0f;
            v = c2 ? (-cm * __sinf(th)) : (cm * __cosf(th));
        }
        sB[idx] = (_Float16)v;
    }
    __syncthreads();

    const int wave = tid >> 5, lane = tid & 31;
    const int ht = wave % 3, wq = wave / 3;              // 3 h-tiles x 2 wt groups
    f8 acc[3] = {};
    for (int kc = 0; kc < 224; kc += 32) {
        h16 a = frag_ld(sA, 224, ht*16, kc);
        for (int j = 0; j < 3; ++j) {
            h16 b = frag_ld(sB, 224, (wq + 2*j)*16, kc);
            acc[j] = wmma16(a, b, acc[j]);
        }
    }
    const int rbase = (lane >> 4) << 3;
    for (int j = 0; j < 3; ++j) {
        int ww = wh*96 + (wq + 2*j)*16 + (lane & 15);
        for (int r = 0; r < 8; ++r) {
            int h = hh*48 + ht*16 + r + rbase;
            y[(size_t)n*NLAT*NLON + (size_t)h*NLON + ww] = acc[j][r];
        }
    }
}

// ---------------------------------------------------------------------------
// Launcher. Workspace: XF (2*97*4096*96 f32 = 291MB) | XG (145MB) | YM (145MB);
// XH aliases XF (dead after k2). ii/jj inputs unused (k = l(l+1)/2 + m).
// ---------------------------------------------------------------------------
extern "C" void kernel_launch(void* const* d_in, const int* in_sizes, int n_in,
                              void* d_out, int out_size, void* d_ws, size_t ws_size,
                              hipStream_t stream) {
    (void)in_sizes; (void)n_in; (void)out_size; (void)ws_size;
    const float* x   = (const float*)d_in[0];
    const float* w   = (const float*)d_in[1];
    const float* pct = (const float*)d_in[2];
    const float* qw  = (const float*)d_in[3];
    float* y = (float*)d_out;

    char* ws = (char*)d_ws;
    const size_t XF_BYTES = (size_t)2*MM*NN*NLAT*4;      // 305,135,616
    const size_t XG_BYTES = (size_t)2*KTOT*NN*4;         // 152,567,808
    float* XF = (float*)ws;
    float* XG = (float*)(ws + XF_BYTES);
    float* YM = (float*)(ws + XF_BYTES + XG_BYTES);
    float* XH = XF;                                      // alias (XF dead after k2)

    k1_dft     <<<dim3(NN, 7, 2),   192, (NLAT*NLON + 16*NLON)*2, stream>>>(x, XF);
    k2_legfwd  <<<dim3(64, MM, 2),  256, (96*96 + 64*96)*2,        stream>>>(XF, pct, qw, XG);
    k3_modegemm<<<dim3(KTOT/2,1,1), 512, K3_LDS,                   stream>>>(XG, w, YM);
    k4_leginv  <<<dim3(64, MM, 2),  256, (64*96 + 96*96)*2,        stream>>>(YM, pct, XH);
    k5_idft    <<<dim3(NN, 2, 2),   192, (48*224 + 96*224)*2,      stream>>>(XH, y);
}
